// LiquidBlock_47407849013298
// MI455X (gfx1250) — matrix operand order repaired and checked by reading, not compiled, not run on hardware
//
#include <hip/hip_runtime.h>
#include <hip/hip_bf16.h>

typedef __attribute__((ext_vector_type(16))) __bf16 v16bf;
typedef __attribute__((ext_vector_type(8)))  float  v8f;

#define TSTEPS 2048
#define BATCH  32

// padded pitches (bf16 elems) and tile counts
#define P0 416   // din0=410 -> 13 k-tiles of 32
#define P1 256   // din1=256 ->  8 k-tiles
#define P2 384   // din2=358 -> 12 k-tiles
#define KT0 13
#define KT1 8
#define KT2 12
#define NT0 10   // dout0=154 -> 10 n-tiles of 16
#define NT1 7    // dout1=102
#define NT2 16   // dout2=256
#define DOUT0 154
#define DOUT1 102
#define DOUT2 256

#define SMEM_BYTES (BATCH * (P0 + P1 + P2) * 2)

static __device__ __forceinline__ unsigned short f2bf_bits(float f) {
  unsigned u = __builtin_bit_cast(unsigned, f);
  unsigned r = u + 0x7FFFu + ((u >> 16) & 1u);   // round-to-nearest-even
  return (unsigned short)(r >> 16);
}
static __device__ __forceinline__ float bf2f(unsigned short b) {
  return __builtin_bit_cast(float, ((unsigned)b) << 16);
}

struct U8 { unsigned int w[8]; };   // 32B carrier for a v16bf fragment

// ---------------- weight packing: W (f32 [dout,din]) * mask -> bf16 WMMA B tiles
// packed tile (kt,nt): 32 lanes x 16 bf16; lane l -> col n = nt*16 + (l&15),
// element j -> k = kt*32 + (j<8 ? (l>>4)*8 + j : 16 + (l>>4)*8 + (j-8))  (B[k][n] = W[n][k])
__global__ void pack_kernel(const float* __restrict__ W, const float* __restrict__ M,
                            unsigned short* __restrict__ out,
                            int din, int dout, int KT, int NT) {
  long i = (long)blockIdx.x * 256 + threadIdx.x;
  long total = (long)KT * NT * 512;
  if (i >= total) return;
  int j = (int)(i & 15);
  int l = (int)((i >> 4) & 31);
  long tile = i >> 9;
  int nt = (int)(tile % NT);
  int kt = (int)(tile / NT);
  int n = nt * 16 + (l & 15);
  int k = kt * 32 + (j < 8 ? (l >> 4) * 8 + j : 16 + (l >> 4) * 8 + (j - 8));
  float v = 0.f;
  if (n < dout && k < din) {
    v = W[(size_t)n * din + k];
    if (M) v *= M[(size_t)n * din + k];
  }
  out[i] = f2bf_bits(v);
}

// ---------------- LayerNorm over D for all (b,t), transpose to xn[t][b][d] bf16
#define LN_TT 8
__global__ void ln_kernel(const float* __restrict__ x, const float* __restrict__ gamma,
                          const float* __restrict__ beta, unsigned short* __restrict__ xn) {
  __shared__ float tile[LN_TT * 256];
  __shared__ float smu[LN_TT], srs[LN_TT];
  int blk = blockIdx.x;
  int b  = blk / (TSTEPS / LN_TT);
  int t0 = (blk % (TSTEPS / LN_TT)) * LN_TT;
  int tid = threadIdx.x;
  for (int i = tid; i < LN_TT * 256; i += 256) {
    int d = i >> 3, tt = i & (LN_TT - 1);
    tile[tt * 256 + d] = x[((size_t)b * 256 + d) * TSTEPS + t0 + tt];
  }
  __syncthreads();
  int wave = tid >> 5, lane = tid & 31;   // 8 waves, one per tt
  {
    float s = 0.f, s2 = 0.f;
    for (int d = lane; d < 256; d += 32) {
      float v = tile[wave * 256 + d];
      s += v; s2 += v * v;
    }
    for (int off = 16; off; off >>= 1) {
      s  += __shfl_xor(s,  off);
      s2 += __shfl_xor(s2, off);
    }
    if (lane == 0) {
      float m = s * (1.f / 256.f);
      smu[wave] = m;
      srs[wave] = rsqrtf(s2 * (1.f / 256.f) - m * m + 1e-5f);
    }
  }
  __syncthreads();
  for (int i = tid; i < LN_TT * 256; i += 256) {
    int tt = i >> 8, d = i & 255;
    float v = (tile[tt * 256 + d] - smu[tt]) * srs[tt] * gamma[d] + beta[d];
    xn[((size_t)(t0 + tt) * BATCH + b) * 256 + d] = f2bf_bits(v);
  }
}

// ---------------- one 16x16 output tile of a CfC cell (all 4 GEMMs fused)
static __device__ __forceinline__ void cfc_tile(
    const unsigned short* __restrict__ xc, int pitch,
    const unsigned short* __restrict__ pw1, const unsigned short* __restrict__ pw2,
    const unsigned short* __restrict__ pwa, const unsigned short* __restrict__ pwb,
    const float* __restrict__ b1, const float* __restrict__ b2,
    const float* __restrict__ ba, const float* __restrict__ bb,
    int mt, int nt, int KT, int NT, int dout, int lane, float out[8]) {
  int n = lane & 15, half = lane >> 4;
  int gn = nt * 16 + n;
  float vb1 = 0.f, vb2 = 0.f, vba = 0.f, vbb = 0.f;
  if (gn < dout) { vb1 = b1[gn]; vb2 = b2[gn]; vba = ba[gn]; vbb = bb[gn]; }
  v8f a1 = {vb1, vb1, vb1, vb1, vb1, vb1, vb1, vb1};
  v8f a2 = {vb2, vb2, vb2, vb2, vb2, vb2, vb2, vb2};
  v8f aa = {vba, vba, vba, vba, vba, vba, vba, vba};
  v8f ab = {vbb, vbb, vbb, vbb, vbb, vbb, vbb, vbb};
  int row = mt * 16 + n;
  const unsigned short* rbase = xc + (size_t)row * pitch + half * 8;
  for (int kt = 0; kt < KT; ++kt) {
    // A fragment: 16-bit A 16x32 layout (two 16B chunks per lane from LDS)
    const unsigned short* ap = rbase + kt * 32;
    U8 ua;
    *(uint4*)&ua.w[0] = *(const uint4*)ap;         // K = kt*32 + half*8 + 0..7
    *(uint4*)&ua.w[4] = *(const uint4*)(ap + 16);  // K = kt*32 + 16 + half*8 + 0..7
    v16bf av = __builtin_bit_cast(v16bf, ua);
    int toff = ((kt * NT + nt) * 32 + lane) * 16;  // ushort units into packed tiles
    if (kt + 1 < KT) {                             // L2 prefetch of next k-tile weights
      __builtin_prefetch(pw1 + toff + NT * 512, 0, 1);
      __builtin_prefetch(pw2 + toff + NT * 512, 0, 1);
    }
    U8 u1, u2, uA, uB;
    *(uint4*)&u1.w[0] = *(const uint4*)(pw1 + toff);
    *(uint4*)&u1.w[4] = *(const uint4*)(pw1 + toff + 8);
    *(uint4*)&u2.w[0] = *(const uint4*)(pw2 + toff);
    *(uint4*)&u2.w[4] = *(const uint4*)(pw2 + toff + 8);
    *(uint4*)&uA.w[0] = *(const uint4*)(pwa + toff);
    *(uint4*)&uA.w[4] = *(const uint4*)(pwa + toff + 8);
    *(uint4*)&uB.w[0] = *(const uint4*)(pwb + toff);
    *(uint4*)&uB.w[4] = *(const uint4*)(pwb + toff + 8);
    a1 = __builtin_amdgcn_wmma_f32_16x16x32_bf16(false, av, false, __builtin_bit_cast(v16bf, u1), (short)0, a1, false, false);
    a2 = __builtin_amdgcn_wmma_f32_16x16x32_bf16(false, av, false, __builtin_bit_cast(v16bf, u2), (short)0, a2, false, false);
    aa = __builtin_amdgcn_wmma_f32_16x16x32_bf16(false, av, false, __builtin_bit_cast(v16bf, uA), (short)0, aa, false, false);
    ab = __builtin_amdgcn_wmma_f32_16x16x32_bf16(false, av, false, __builtin_bit_cast(v16bf, uB), (short)0, ab, false, false);
  }
  for (int r = 0; r < 8; ++r) {
    float f1 = tanhf(a1[r]);
    float f2 = tanhf(a2[r]);
    float ti = 1.f / (1.f + __expf(-(aa[r] + ab[r])));
    out[r] = f1 + ti * (f2 - f1);
  }
}

struct ScanArgs {
  const unsigned short* xn;
  const unsigned short* w[12];   // [layer*4 + {W1,W2,Wa,Wb}] packed tiles
  const float* bias[12];         // [layer*4 + {b1,b2,ba,bb}]
  float* y;
  float* hT;
};

// ---------------- persistent sequential scan: 1 workgroup, 16 waves, loops T
__global__ void __launch_bounds__(512) scan_kernel(ScanArgs a) {
  extern __shared__ unsigned short smem[];
  unsigned short* xc0 = smem;                    // [32][P0]: xn_t | h0_prev
  unsigned short* xc1 = xc0 + BATCH * P0;        // [32][P1]: h0_new | h1_prev
  unsigned short* xc2 = xc1 + BATCH * P1;        // [32][P2]: h1_new | h2_prev
  int tid = threadIdx.x;
  int lane = tid & 31, wave = tid >> 5;

  for (int i = tid; i < BATCH * P0; i += 512) xc0[i] = 0;
  for (int i = tid; i < BATCH * P1; i += 512) xc1[i] = 0;
  for (int i = tid; i < BATCH * P2; i += 512) xc2[i] = 0;
  __syncthreads();

  float res[2][8];
  int   tl[2];

  for (int t = 0; t < TSTEPS; ++t) {
    // stage normalized input for this step into xc0[:, 0:256]
    const uint4* src = (const uint4*)(a.xn + (size_t)t * BATCH * 256);
    for (int i = tid; i < BATCH * 256 / 8; i += 512) {
      int b = i >> 5, c = i & 31;
      *(uint4*)&xc0[b * P0 + c * 8] = src[i];
    }
    __syncthreads();

    // ---- layer 0: concat(xn_t, h0) -> h0_new
    int cnt = 0;
    for (int tile = wave; tile < 2 * NT0; tile += 16) {
      cfc_tile(xc0, P0, a.w[0], a.w[1], a.w[2], a.w[3],
               a.bias[0], a.bias[1], a.bias[2], a.bias[3],
               tile / NT0, tile % NT0, KT0, NT0, DOUT0, lane, res[cnt]);
      tl[cnt++] = tile;
    }
    __syncthreads();
    for (int c = 0; c < cnt; ++c) {
      int mt = tl[c] / NT0, nt = tl[c] % NT0;
      int gn = nt * 16 + (lane & 15), half = lane >> 4;
      if (gn < DOUT0)
        for (int r = 0; r < 8; ++r) {
          int m = mt * 16 + r + 8 * half;
          unsigned short bv = f2bf_bits(res[c][r]);
          xc1[m * P1 + gn] = bv;          // layer1 input
          xc0[m * P0 + 256 + gn] = bv;    // carry for next step
        }
    }
    __syncthreads();

    // ---- layer 1: concat(h0_new, h1) -> h1_new
    cnt = 0;
    for (int tile = wave; tile < 2 * NT1; tile += 16) {
      cfc_tile(xc1, P1, a.w[4], a.w[5], a.w[6], a.w[7],
               a.bias[4], a.bias[5], a.bias[6], a.bias[7],
               tile / NT1, tile % NT1, KT1, NT1, DOUT1, lane, res[cnt]);
      tl[cnt++] = tile;
    }
    __syncthreads();
    for (int c = 0; c < cnt; ++c) {
      int mt = tl[c] / NT1, nt = tl[c] % NT1;
      int gn = nt * 16 + (lane & 15), half = lane >> 4;
      if (gn < DOUT1)
        for (int r = 0; r < 8; ++r) {
          int m = mt * 16 + r + 8 * half;
          unsigned short bv = f2bf_bits(res[c][r]);
          xc2[m * P2 + gn] = bv;          // layer2 input
          xc1[m * P1 + 154 + gn] = bv;    // carry
        }
    }
    __syncthreads();

    // ---- layer 2: concat(h1_new, h2) -> h2_new == y[:, :, t]
    cnt = 0;
    for (int tile = wave; tile < 2 * NT2; tile += 16) {
      cfc_tile(xc2, P2, a.w[8], a.w[9], a.w[10], a.w[11],
               a.bias[8], a.bias[9], a.bias[10], a.bias[11],
               tile / NT2, tile % NT2, KT2, NT2, DOUT2, lane, res[cnt]);
      tl[cnt++] = tile;
    }
    __syncthreads();
    for (int c = 0; c < cnt; ++c) {
      int mt = tl[c] / NT2, nt = tl[c] % NT2;
      int gn = nt * 16 + (lane & 15), half = lane >> 4;
      for (int r = 0; r < 8; ++r) {
        int m = mt * 16 + r + 8 * half;
        xc2[m * P2 + 102 + gn] = f2bf_bits(res[c][r]);       // carry
        a.y[((size_t)m * 256 + gn) * TSTEPS + t] = res[c][r];
      }
    }
    __syncthreads();
  }

  // final hidden state: concat(h0[154], h1[102], h2[256]) per batch row
  for (int i = tid; i < BATCH * 512; i += 512) {
    int b = i >> 9, j = i & 511;
    float v;
    if (j < 154)      v = bf2f(xc0[b * P0 + 256 + j]);
    else if (j < 256) v = bf2f(xc1[b * P1 + 154 + (j - 154)]);
    else              v = bf2f(xc2[b * P2 + 102 + (j - 256)]);
    a.hT[i] = v;
  }
}

extern "C" void kernel_launch(void* const* d_in, const int* in_sizes, int n_in,
                              void* d_out, int out_size, void* d_ws, size_t ws_size,
                              hipStream_t stream) {
  (void)in_sizes; (void)n_in; (void)out_size; (void)ws_size;
  const float* x     = (const float*)d_in[0];
  const float* gamma = (const float*)d_in[1];
  const float* beta  = (const float*)d_in[2];

  // workspace: xn[T][B][256] bf16, then 12 packed weight blocks
  unsigned short* ws = (unsigned short*)d_ws;
  unsigned short* xn = ws;
  size_t off = (size_t)TSTEPS * BATCH * 256;

  const int KT[3]  = {KT0, KT1, KT2};
  const int NT[3]  = {NT0, NT1, NT2};
  const int din[3] = {410, 256, 358};
  const int dou[3] = {DOUT0, DOUT1, DOUT2};

  ScanArgs sa;
  sa.xn = xn;
  for (int l = 0; l < 3; ++l) {
    const float* mask = (const float*)d_in[3 + 9 * l + 8];
    for (int wi = 0; wi < 4; ++wi) {
      unsigned short* dst = ws + off;
      sa.w[l * 4 + wi] = dst;
      sa.bias[l * 4 + wi] = (const float*)d_in[3 + 9 * l + 4 + wi];
      long total = (long)KT[l] * NT[l] * 512;
      off += (size_t)total;
      const float* W = (const float*)d_in[3 + 9 * l + wi];
      const float* M = (wi < 2) ? mask : nullptr;   // mask applies to W1/W2 only
      pack_kernel<<<(int)((total + 255) / 256), 256, 0, stream>>>(W, M, dst, din[l], dou[l], KT[l], NT[l]);
    }
  }

  ln_kernel<<<(BATCH * TSTEPS) / LN_TT, 256, 0, stream>>>(x, gamma, beta, xn);

  sa.y  = (float*)d_out;
  sa.hT = sa.y + (size_t)BATCH * 256 * TSTEPS;

  (void)hipFuncSetAttribute((const void*)scan_kernel,
                            hipFuncAttributeMaxDynamicSharedMemorySize, SMEM_BYTES);
  scan_kernel<<<1, 512, SMEM_BYTES, stream>>>(sa);
}
